// SkeletonLSTM_60017872994931
// MI455X (gfx1250) — matrix operand order repaired and checked by reading, not compiled, not run on hardware
//
#include <hip/hip_runtime.h>
#include <hip/hip_bf16.h>

// Problem constants
#define BB 256
#define TT 196
#define FF 263
#define HH 1024
#define KX 2048     // concat K for gates GEMM: [enc(512)|text(512)|h(1024)]
#define F_PAD 288   // frame K padded to multiple of 32 (>= 263)
#define N2_PAD 272  // lin2 output N padded to multiple of 16 (>= 263)

typedef __bf16 bf16_t;
typedef __attribute__((ext_vector_type(16))) bf16_t v16bf;
typedef __attribute__((ext_vector_type(8)))  float  v8f;

// ---- workspace layout (bytes) ----
// wc      : bf16 [4096][2048]   packed [w_ih | w_hh]
// l1w     : bf16 [512][288]     lin1_w zero-padded in K
// l2w     : bf16 [272][1024]    lin2_w zero-padded in N(rows)
// xh      : bf16 [256][2048]    activation buffer [enc | text | h]
// cstate  : f32  [256][1024]
// frame   : f32  [256][263]
// framebf : bf16 [256][288]     frame zero-padded in K
static constexpr size_t WC_OFF  = 0;
static constexpr size_t L1W_OFF = WC_OFF  + (size_t)4096 * 2048 * 2;
static constexpr size_t L2W_OFF = L1W_OFF + (size_t)512 * F_PAD * 2;
static constexpr size_t XH_OFF  = L2W_OFF + (size_t)N2_PAD * HH * 2;
static constexpr size_t C_OFF   = XH_OFF  + (size_t)BB * KX * 2;
static constexpr size_t FR_OFF  = C_OFF   + (size_t)BB * HH * 4;
static constexpr size_t FRB_OFF = FR_OFF  + (size_t)BB * FF * 4;

// ---------------- WMMA fragment loaders (wave32 layouts, ISA 7.12.2) -------
// A 16x32 bf16: lane l holds row m0+(l&15); elements 0..7 = K k0+koff+0..7,
// elements 8..15 = K k0+koff+16..23, koff = (l>=16) ? 8 : 0.
__device__ inline v16bf load_frag_A(const bf16_t* X, int ldx, int m0, int k0) {
    int lane = threadIdx.x & 31;
    int m    = m0 + (lane & 15);
    int koff = (lane & 16) ? 8 : 0;
    const bf16_t* p = X + (size_t)m * ldx + k0 + koff;
    union { v16bf v; uint4 q[2]; } u;
    u.q[0] = *(const uint4*)(p);
    u.q[1] = *(const uint4*)(p + 16);
    return u.v;
}

// B 32x16 bf16 for D = A * W^T: lane l holds column n0+(l&15) = row of W;
// elements 0..7 = K k0+koff+0..7, elements 8..15 = K k0+koff+16..23, koff = (l>=16) ? 8 : 0.
__device__ inline v16bf load_frag_B(const bf16_t* W, int ldw, int n0, int k0) {
    int lane = threadIdx.x & 31;
    int n    = n0 + (lane & 15);
    int koff = (lane & 16) ? 8 : 0;
    const bf16_t* p = W + (size_t)n * ldw + k0 + koff;
    union { v16bf v; uint4 q[2]; } u;
    u.q[0] = *(const uint4*)(p);
    u.q[1] = *(const uint4*)(p + 16);
    return u.v;
}

__device__ inline float sigmoidf_(float x) { return 1.0f / (1.0f + __expf(-x)); }

// ------------------------------- prep kernels ------------------------------
__global__ void slstm_prep_wc(const float* __restrict__ w_ih,
                              const float* __restrict__ w_hh,
                              bf16_t* __restrict__ wc) {
    size_t total = (size_t)4 * HH * KX;
    for (size_t i = (size_t)blockIdx.x * blockDim.x + threadIdx.x; i < total;
         i += (size_t)gridDim.x * blockDim.x) {
        size_t n = i / KX, k = i % KX;
        float v = (k < HH) ? w_ih[n * HH + k] : w_hh[n * HH + (k - HH)];
        wc[i] = (bf16_t)v;
    }
}

__global__ void slstm_prep_lin1(const float* __restrict__ lin1_w,
                                bf16_t* __restrict__ l1w) {
    size_t total = (size_t)512 * F_PAD;
    for (size_t i = (size_t)blockIdx.x * blockDim.x + threadIdx.x; i < total;
         i += (size_t)gridDim.x * blockDim.x) {
        size_t n = i / F_PAD, k = i % F_PAD;
        l1w[i] = (bf16_t)((k < FF) ? lin1_w[n * FF + k] : 0.0f);
    }
}

__global__ void slstm_prep_lin2(const float* __restrict__ lin2_w,
                                bf16_t* __restrict__ l2w) {
    size_t total = (size_t)N2_PAD * HH;
    for (size_t i = (size_t)blockIdx.x * blockDim.x + threadIdx.x; i < total;
         i += (size_t)gridDim.x * blockDim.x) {
        size_t n = i / HH, k = i % HH;
        l2w[i] = (bf16_t)((n < FF) ? lin2_w[n * HH + k] : 0.0f);
    }
}

__global__ void slstm_init_state(const float* __restrict__ motions,
                                 float* __restrict__ cstate,
                                 bf16_t* __restrict__ xh,
                                 float* __restrict__ frame,
                                 bf16_t* __restrict__ framebf) {
    size_t i = (size_t)blockIdx.x * blockDim.x + threadIdx.x;
    if (i < (size_t)BB * HH) {
        cstate[i] = 0.0f;
        size_t b = i / HH, j = i % HH;
        xh[b * KX + HH + j] = (bf16_t)0.0f;     // h0 = 0
    }
    if (i < (size_t)BB * F_PAD) {
        size_t b = i / F_PAD, f = i % F_PAD;
        float v = (f < FF) ? motions[b * (size_t)TT * FF + f] : 0.0f;
        framebf[i] = (bf16_t)v;
        if (f < FF) frame[b * FF + f] = v;
    }
}

// text_emb = text_embedding @ lin_text_w.T + b  (one-time, fp32 scalar)
__global__ void slstm_text_emb(const float* __restrict__ text_embedding,
                               const float* __restrict__ lin_text_w,
                               const float* __restrict__ lin_text_b,
                               bf16_t* __restrict__ xh) {
    size_t i = (size_t)blockIdx.x * blockDim.x + threadIdx.x;
    if (i >= (size_t)BB * 512) return;
    size_t b = i / 512, j = i % 512;
    const float* x = text_embedding + b * 768;
    const float* w = lin_text_w + j * 768;
    float acc = lin_text_b[j];
    for (int k = 0; k < 768; ++k) acc = fmaf(x[k], w[k], acc);
    xh[b * KX + 512 + j] = (bf16_t)acc;
}

// ------------------------------- step kernels ------------------------------
// enc = frame @ lin1_w.T + lin1_b  -> xh[:, 0:512] (bf16)
// grid (8,16), 128 threads (4 waves 2x2), block tile 32(M) x 32(N)
__global__ void slstm_enc(const bf16_t* __restrict__ framebf,
                          const bf16_t* __restrict__ l1w,
                          const float* __restrict__ lin1_b,
                          bf16_t* __restrict__ xh) {
    int lane = threadIdx.x & 31, wid = threadIdx.x >> 5;
    int m0 = blockIdx.x * 32 + (wid >> 1) * 16;
    int n0 = blockIdx.y * 32 + (wid & 1) * 16;
    int n  = n0 + (lane & 15);
    v8f acc;
    float bias = lin1_b[n];
    #pragma unroll
    for (int r = 0; r < 8; ++r) acc[r] = bias;
    for (int k = 0; k < F_PAD; k += 32) {
        v16bf a = load_frag_A(framebf, F_PAD, m0, k);
        v16bf b = load_frag_B(l1w, F_PAD, n0, k);
        acc = __builtin_amdgcn_wmma_f32_16x16x32_bf16(false, a, false, b,
                                                      (short)0, acc, false, false);
    }
    int hi = (lane & 16) ? 8 : 0;
    #pragma unroll
    for (int r = 0; r < 8; ++r) {
        int m = m0 + r + hi;
        xh[(size_t)m * KX + n] = (bf16_t)acc[r];
    }
}

// gates + activations + c/h update.
// gate_q[m,j] = sum_k xh[m,k]*wc[q*H+j,k] + b_ih[q*H+j] + b_hh[q*H+j]
// grid (8,32), 128 threads (4 waves 2x2), block tile 32(M) x 32(J)
__global__ void slstm_gates(const bf16_t* __restrict__ xh_ro,
                            const bf16_t* __restrict__ wc,
                            const float* __restrict__ b_ih,
                            const float* __restrict__ b_hh,
                            float* __restrict__ cstate,
                            bf16_t* __restrict__ xh_h) {
    int lane = threadIdx.x & 31, wid = threadIdx.x >> 5;
    int m0 = blockIdx.x * 32 + (wid >> 1) * 16;
    int j0 = blockIdx.y * 32 + (wid & 1) * 16;
    int j  = j0 + (lane & 15);
    v8f acc[4];
    #pragma unroll
    for (int q = 0; q < 4; ++q) {
        float bias = b_ih[q * HH + j] + b_hh[q * HH + j];
        #pragma unroll
        for (int r = 0; r < 8; ++r) acc[q][r] = bias;
    }
    for (int k = 0; k < KX; k += 32) {
        v16bf a = load_frag_A(xh_ro, KX, m0, k);
        #pragma unroll
        for (int q = 0; q < 4; ++q) {
            v16bf b = load_frag_B(wc, KX, q * HH + j0, k);
            acc[q] = __builtin_amdgcn_wmma_f32_16x16x32_bf16(false, a, false, b,
                                                             (short)0, acc[q],
                                                             false, false);
        }
    }
    int hi = (lane & 16) ? 8 : 0;
    #pragma unroll
    for (int r = 0; r < 8; ++r) {
        int m = m0 + r + hi;
        float iv = sigmoidf_(acc[0][r]);
        float fv = sigmoidf_(acc[1][r]);
        float gv = tanhf(acc[2][r]);
        float ov = sigmoidf_(acc[3][r]);
        size_t ci = (size_t)m * HH + j;
        float cn = fv * cstate[ci] + iv * gv;
        cstate[ci] = cn;
        float hv = ov * tanhf(cn);
        xh_h[(size_t)m * KX + HH + j] = (bf16_t)hv;
    }
}

// out = h @ lin2_w.T + lin2_b ; frame += out ; store d_out[:, t, :]
// grid (8,17), 64 threads (2 waves), block tile 32(M) x 16(N)
__global__ void slstm_out(const bf16_t* __restrict__ xh,
                          const bf16_t* __restrict__ l2w,
                          const float* __restrict__ lin2_b,
                          float* __restrict__ frame,
                          bf16_t* __restrict__ framebf,
                          float* __restrict__ out, int t) {
    int lane = threadIdx.x & 31, wid = threadIdx.x >> 5;
    int m0 = blockIdx.x * 32 + wid * 16;
    int n0 = blockIdx.y * 16;
    int n  = n0 + (lane & 15);
    v8f acc;
    float bias = (n < FF) ? lin2_b[n] : 0.0f;
    #pragma unroll
    for (int r = 0; r < 8; ++r) acc[r] = bias;
    for (int k = 0; k < HH; k += 32) {
        v16bf a = load_frag_A(xh, KX, m0, HH + k);   // h lives at xh[:,1024:2048]
        v16bf b = load_frag_B(l2w, HH, n0, k);
        acc = __builtin_amdgcn_wmma_f32_16x16x32_bf16(false, a, false, b,
                                                      (short)0, acc, false, false);
    }
    int hi = (lane & 16) ? 8 : 0;
    if (n < FF) {
        #pragma unroll
        for (int r = 0; r < 8; ++r) {
            int m = m0 + r + hi;
            size_t fi = (size_t)m * FF + n;
            float nf = frame[fi] + acc[r];
            frame[fi] = nf;
            framebf[(size_t)m * F_PAD + n] = (bf16_t)nf;
            out[(size_t)m * TT * FF + (size_t)t * FF + n] = nf;
        }
    }
}

// ------------------------------- launcher ----------------------------------
extern "C" void kernel_launch(void* const* d_in, const int* in_sizes, int n_in,
                              void* d_out, int out_size, void* d_ws, size_t ws_size,
                              hipStream_t stream) {
    const float* motions        = (const float*)d_in[0];
    const float* text_embedding = (const float*)d_in[1];
    const float* lin_text_w     = (const float*)d_in[2];
    const float* lin_text_b     = (const float*)d_in[3];
    const float* lin1_w         = (const float*)d_in[4];
    const float* lin1_b         = (const float*)d_in[5];
    const float* w_ih           = (const float*)d_in[6];
    const float* w_hh           = (const float*)d_in[7];
    const float* b_ih           = (const float*)d_in[8];
    const float* b_hh           = (const float*)d_in[9];
    const float* lin2_w         = (const float*)d_in[10];
    const float* lin2_b         = (const float*)d_in[11];
    float* out = (float*)d_out;

    char* ws = (char*)d_ws;
    bf16_t* wc      = (bf16_t*)(ws + WC_OFF);
    bf16_t* l1w     = (bf16_t*)(ws + L1W_OFF);
    bf16_t* l2w     = (bf16_t*)(ws + L2W_OFF);
    bf16_t* xh      = (bf16_t*)(ws + XH_OFF);
    float*  cstate  = (float*) (ws + C_OFF);
    float*  frame   = (float*) (ws + FR_OFF);
    bf16_t* framebf = (bf16_t*)(ws + FRB_OFF);

    // one-time prep (re-run every call for determinism)
    slstm_prep_wc<<<4096, 256, 0, stream>>>(w_ih, w_hh, wc);
    slstm_prep_lin1<<<(512 * F_PAD + 255) / 256, 256, 0, stream>>>(lin1_w, l1w);
    slstm_prep_lin2<<<(N2_PAD * HH + 255) / 256, 256, 0, stream>>>(lin2_w, l2w);
    slstm_init_state<<<(BB * HH + 255) / 256, 256, 0, stream>>>(motions, cstate, xh,
                                                                frame, framebf);
    slstm_text_emb<<<(BB * 512 + 255) / 256, 256, 0, stream>>>(text_embedding,
                                                               lin_text_w,
                                                               lin_text_b, xh);
    // sequential recurrence
    for (int t = 0; t < TT; ++t) {
        slstm_enc<<<dim3(BB / 32, 512 / 32), 128, 0, stream>>>(framebf, l1w,
                                                               lin1_b, xh);
        slstm_gates<<<dim3(BB / 32, HH / 32), 128, 0, stream>>>(xh, wc, b_ih,
                                                                b_hh, cstate, xh);
        slstm_out<<<dim3(BB / 32, N2_PAD / 16), 64, 0, stream>>>(xh, l2w, lin2_b,
                                                                 frame, framebf,
                                                                 out, t);
    }
}
